// digitalFilm_G_85959475462874
// MI455X (gfx1250) — matrix operand order repaired and checked by reading, not compiled, not run on hardware
//
#include <hip/hip_runtime.h>
#include <hip/hip_bf16.h>
#include <math.h>

typedef __attribute__((ext_vector_type(16))) _Float16 v16h;
typedef __attribute__((ext_vector_type(8)))  _Float16 v8h;
typedef __attribute__((ext_vector_type(4)))  _Float16 v4h;
typedef __attribute__((ext_vector_type(8)))  float    v8f;
typedef _Float16 f16;

union AFrag {
  v16h v;
  v8h  r[2];
  v4h  q[4];
};

// ---------------------------------------------------------------------------
// x (N,3,H,W) f32 -> NHWC f16 with C=4 (rgb0) and 1-pixel zero halo.
// Layout: [N][H+2][W+2][4]
// ---------------------------------------------------------------------------
__global__ void pack_x_kernel(const float* __restrict__ x, f16* __restrict__ xh,
                              int N, int H, int W) {
  const int Hp = H + 2, Wp = W + 2;
  int idx = blockIdx.x * blockDim.x + threadIdx.x;
  if (idx >= N * Hp * Wp) return;
  const int n  = idx / (Hp * Wp);
  const int rp = idx - n * (Hp * Wp);
  const int yp = rp / Wp;
  const int xp = rp - yp * Wp;
  f16 v0 = (f16)0.0f, v1 = (f16)0.0f, v2 = (f16)0.0f;
  if (yp >= 1 && yp <= H && xp >= 1 && xp <= W) {
    const float* xb = x + (size_t)n * 3 * H * W + (size_t)(yp - 1) * W + (xp - 1);
    v0 = (f16)xb[0];
    v1 = (f16)xb[(size_t)H * W];
    v2 = (f16)xb[2 * (size_t)H * W];
  }
  v4h o = { v0, v1, v2, (f16)0.0f };
  *(v4h*)(xh + (size_t)idx * 4) = o;
}

// ---------------------------------------------------------------------------
// Zero the halo border pixels of an NHWC activation [N][Hout+2][Wout+2][C].
// ---------------------------------------------------------------------------
__global__ void zero_halo_kernel(f16* __restrict__ act, int N, int Hout, int Wout,
                                 int C) {
  const int Hp = Hout + 2, Wp = Wout + 2;
  int idx = blockIdx.x * blockDim.x + threadIdx.x;
  if (idx >= N * Hp * Wp) return;
  const int n  = idx / (Hp * Wp);
  const int rp = idx - n * (Hp * Wp);
  const int yp = rp / Wp;
  const int xp = rp - yp * Wp;
  if (yp == 0 || yp == Hp - 1 || xp == 0 || xp == Wp - 1) {
    f16* p = act + (size_t)idx * C;
    for (int c = 0; c < C; ++c) p[c] = (f16)0.0f;
  }
}

// ---------------------------------------------------------------------------
// Pack conv weights (Cout, CinReal, 3, 3) f32 -> WMMA B-panel f16 layout.
// GEMM K ordering: k = tap*Cpad + cin  (tap = kh*3+kw), zero for padding.
// Per (ctile, ktile) 32x16 block (512 f16): [khalf(2)][col(16)][q(16)],
// so lane L loads 16 consecutive f16 at ((L/16)*16 + L%16)*16 — matching
// the contiguous K order also used for the A fragment (lanes 0-15: K=0..15, lanes 16-31: K=16..31,
// VGPR j = K pair 2j,2j+1).
// ---------------------------------------------------------------------------
__global__ void pack_w_kernel(const float* __restrict__ w, f16* __restrict__ packed,
                              int Cout, int CinReal, int Cpad, int Kpad) {
  int idx = blockIdx.x * blockDim.x + threadIdx.x;
  if (idx >= Cout * Kpad) return;
  const int c = idx / Kpad;
  const int k = idx - c * Kpad;
  const int tap = k / Cpad;
  const int cin = k - tap * Cpad;
  float v = 0.0f;
  if (tap < 9 && cin < CinReal) v = w[((size_t)c * CinReal + cin) * 9 + tap];
  const int ctile = c >> 4, col = c & 15;
  const int ktile = k >> 5, kk = k & 31, khalf = kk >> 4, q = kk & 15;
  const int KT = Kpad >> 5;
  size_t off = ((((size_t)ctile * KT + ktile) * 2 + khalf) * 16 + col) * 16 + q;
  packed[off] = (f16)v;
}

// ---------------------------------------------------------------------------
// Implicit-GEMM conv3x3 (pad=1) + bias + ReLU via v_wmma_f32_16x16x32_f16.
// NHWC activations with 1-px zero halo:
//   in : [N][Hin+2][Win+2][CIN]     out: [N][Hout+2][Wout+2][NT*16]
// One wave: 16 consecutive ox (one row, wave-uniform n/oy) x all NT col-tiles.
// A fragment (contiguous K order, as in the B panel): lane L (m=L%16, half=L/16) needs two
// contiguous 8-half K-runs: K0+half*16 and K0+half*16+8.  With channels-inner
// K ordering each run is 1 contiguous 16B load (CIN>=16) or 2x 8B (CIN==4).
// K-padding taps are clamped (B panel is zero there, so A values are don't-care).
// ---------------------------------------------------------------------------
template <int CIN, int NT, int STRIDE>
__global__ __launch_bounds__(128)
void conv3x3_wmma(const f16* __restrict__ in, const f16* __restrict__ wpk,
                  const float* __restrict__ bias, f16* __restrict__ out,
                  int N, int Hin, int Win) {
  const int Hout = (STRIDE == 1) ? Hin : (Hin >> 1);
  const int Wout = (STRIDE == 1) ? Win : (Win >> 1);
  const int lane = threadIdx.x & 31;
  const int wave = threadIdx.x >> 5;
  const int tileM = blockIdx.x * 4 + wave;
  const int tilesPerRow = Wout >> 4;
  const int rowId = tileM / tilesPerRow;          // n*Hout + oy (wave-uniform)
  if (rowId >= N * Hout) return;
  const int n   = rowId / Hout;
  const int oy  = rowId - n * Hout;
  const int ox0 = (tileM - rowId * tilesPerRow) << 4;

  const int half = lane >> 4;
  const int col  = lane & 15;
  const int Wp   = Win + 2;
  const int ix   = (ox0 + col) * STRIDE;          // + kw gives padded x coord
  const size_t rowPix = ((size_t)n * (Hin + 2) + (size_t)oy * STRIDE) * Wp + ix;

  constexpr int Ktot = 9 * CIN;
  constexpr int KT   = (Ktot + 31) / 32;

  v8f acc[NT] = {};

  for (int ks = 0; ks < KT; ++ks) {
    AFrag a;
#pragma unroll
    for (int r = 0; r < 2; ++r) {
      const int k = ks * 32 + half * 16 + r * 8;
      if (CIN >= 8) {
        int tap = k / CIN;
        if (tap > 8) tap = 8;                      // padding: B is zero there
        const int cin0 = k & (CIN - 1);
        const int kh = tap / 3, kw = tap - (tap / 3) * 3;
        a.r[r] = *(const v8h*)(in + (rowPix + (size_t)kh * Wp + kw) * CIN + cin0);
      } else {  // CIN == 4: run spans two adjacent taps
#pragma unroll
        for (int q = 0; q < 2; ++q) {
          int tap = (k >> 2) + q;
          if (tap > 8) tap = 8;
          const int kh = tap / 3, kw = tap - (tap / 3) * 3;
          a.q[r * 2 + q] = *(const v4h*)(in + (rowPix + (size_t)kh * Wp + kw) * 4);
        }
      }
    }
#pragma unroll
    for (int t = 0; t < NT; ++t) {
      const v16h* bp = (const v16h*)(wpk + ((size_t)(t * KT + ks)) * 512);
      v16h b = bp[half * 16 + col];
      acc[t] = __builtin_amdgcn_wmma_f32_16x16x32_f16(
          false, a.v, false, b, (short)0, acc[t], false, false);
    }
  }

  // epilogue: bias + relu, f16 NHWC-halo store
  // D layout: lane L -> col = L%16 (out channel), VGPR v -> row m = v + 8*half
  constexpr int Cout = NT * 16;
  const int Wpo = Wout + 2;
  const size_t outRow = ((size_t)n * (Hout + 2) + (oy + 1)) * Wpo + 1 + ox0;
#pragma unroll
  for (int t = 0; t < NT; ++t) {
    const int c = t * 16 + col;
    const float bs = bias[c];
#pragma unroll
    for (int v = 0; v < 8; ++v) {
      const int md = v + 8 * half;
      float val = acc[t][v] + bs;
      val = val > 0.0f ? val : 0.0f;
      out[(outRow + md) * Cout + c] = (f16)val;
    }
  }
}

// ---------------------------------------------------------------------------
// zero a small f32 region
// ---------------------------------------------------------------------------
__global__ void zero_f32_kernel(float* __restrict__ p, int n) {
  int i = blockIdx.x * blockDim.x + threadIdx.x;
  if (i < n) p[i] = 0.0f;
}

// ---------------------------------------------------------------------------
// NHWC global-sum pool (one block per (n,row); threads = channels; atomics).
// Mean scaling folded into the consuming FC.
// ---------------------------------------------------------------------------
__global__ void pool_rows_kernel(const f16* __restrict__ act, float* __restrict__ feat,
                                 int Hout, int Wout, int C) {
  const int row = blockIdx.x;
  const int n = row / Hout, y = row - n * Hout;
  const int c = threadIdx.x;
  const int Wp = Wout + 2;
  const f16* p = act + (((size_t)n * (Hout + 2) + (y + 1)) * Wp + 1) * C + c;
  float s = 0.0f;
  for (int x = 0; x < Wout; ++x) s += (float)p[(size_t)x * C];
  atomicAdd(&feat[n * C + c], s);
}

// ---------------------------------------------------------------------------
// Tiny FC: out[n,o] = act(sum_i feat[n,i]*inScale * W[o,i] + b[o])
// ---------------------------------------------------------------------------
__global__ void fc_kernel(const float* __restrict__ feat, const float* __restrict__ W,
                          const float* __restrict__ b, float* __restrict__ out,
                          int N, int I, int O, int relu, float inScale) {
  int idx = blockIdx.x * blockDim.x + threadIdx.x;
  if (idx >= N * O) return;
  int n = idx / O, o = idx - (idx / O) * O;
  float s = b[o];
  for (int i = 0; i < I; ++i) s += feat[n * I + i] * inScale * W[o * I + i];
  if (relu) s = s > 0.0f ? s : 0.0f;
  out[idx] = s;
}

// ---------------------------------------------------------------------------
// fused = clip(weight3d @ luts + identity3d, 0, 1)
// ---------------------------------------------------------------------------
__global__ void fuse_luts_kernel(const float* __restrict__ w3d,
                                 const float* __restrict__ luts,
                                 float* __restrict__ fused,
                                 int B, int NB, int D) {
  const int V = 3 * D * D * D;
  int idx = blockIdx.x * blockDim.x + threadIdx.x;
  if (idx >= B * V) return;
  const int bimg = idx / V;
  const int e = idx - bimg * V;
  float s = 0.0f;
  for (int nb = 0; nb < NB; ++nb)
    s += w3d[bimg * NB + nb] * luts[(size_t)nb * V + e];
  const int D3 = D * D * D;
  const int c  = e / D3;
  const int r3 = e - c * D3;
  const int bz = r3 / (D * D);
  const int r2 = r3 - bz * (D * D);
  const int gy = r2 / D;
  const int rx = r2 - gy * D;
  const int st = (c == 0) ? rx : (c == 1) ? gy : bz;
  s += (float)st / (float)(D - 1);
  s = fminf(fmaxf(s, 0.0f), 1.0f);
  fused[idx] = s;
}

// ---------------------------------------------------------------------------
// TV + monotonicity regularizers over fused (B,3,D,D,D)
// ---------------------------------------------------------------------------
__global__ void tv_mn_kernel(const float* __restrict__ fused,
                             float* __restrict__ tv_out, float* __restrict__ mn_out,
                             int B, int D) {
  const int D1 = D - 1;
  const int cnt = B * 3 * D * D * D1;
  const float scale = 1.0f / (float)cnt;
  const int D3 = D * D * D;
  float tv = 0.0f, mn = 0.0f;
  for (int idx = blockIdx.x * blockDim.x + threadIdx.x; idx < cnt;
       idx += gridDim.x * blockDim.x) {
    {  // dr : last axis
      int i = idx;
      const int xr = i % D1; i /= D1;
      const int y  = i % D;  i /= D;
      const int z  = i % D;  i /= D;
      const int c  = i % 3;  const int bi = i / 3;
      const float* L = fused + (size_t)(bi * 3 + c) * D3 + ((size_t)z * D + y) * D;
      const float d = L[xr] - L[xr + 1];
      const float w = (xr == 0 || xr == D - 2) ? 2.0f : 1.0f;
      tv += d * d * w;
      mn += d > 0.0f ? d : 0.0f;
    }
    {  // dg : y axis
      int i = idx;
      const int xr = i % D;  i /= D;
      const int y  = i % D1; i /= D1;
      const int z  = i % D;  i /= D;
      const int c  = i % 3;  const int bi = i / 3;
      const float* L = fused + (size_t)(bi * 3 + c) * D3 + (size_t)z * D * D;
      const float d = L[y * D + xr] - L[(y + 1) * D + xr];
      const float w = (y == 0 || y == D - 2) ? 2.0f : 1.0f;
      tv += d * d * w;
      mn += d > 0.0f ? d : 0.0f;
    }
    {  // db : z axis
      int i = idx;
      const int xr = i % D;  i /= D;
      const int y  = i % D;  i /= D;
      const int z  = i % D1; i /= D1;
      const int c  = i % 3;  const int bi = i / 3;
      const float* L = fused + (size_t)(bi * 3 + c) * D3;
      const float d = L[(z * D + y) * D + xr] - L[((z + 1) * D + y) * D + xr];
      const float w = (z == 0 || z == D - 2) ? 2.0f : 1.0f;
      tv += d * d * w;
      mn += d > 0.0f ? d : 0.0f;
    }
  }
  __shared__ float stv[256], smn[256];
  stv[threadIdx.x] = tv * scale;
  smn[threadIdx.x] = mn * scale;
  __syncthreads();
  for (int o = 128; o > 0; o >>= 1) {
    if ((int)threadIdx.x < o) {
      stv[threadIdx.x] += stv[threadIdx.x + o];
      smn[threadIdx.x] += smn[threadIdx.x + o];
    }
    __syncthreads();
  }
  if (threadIdx.x == 0) {
    atomicAdd(tv_out, stv[0]);
    atomicAdd(mn_out, smn[0]);
  }
}

// ---------------------------------------------------------------------------
// Trilinear LUT apply + sigmoid blend + clip (bandwidth bound, ~100 MB)
// ---------------------------------------------------------------------------
__global__ void trilerp_blend_kernel(const float* __restrict__ x,
                                     const float* __restrict__ fused,
                                     const float* __restrict__ logit,
                                     float* __restrict__ out,
                                     int B, int HW, int D) {
  int idx = blockIdx.x * blockDim.x + threadIdx.x;
  if (idx >= B * HW) return;
  const int n = idx / HW;
  const int pix = idx - n * HW;
  const float* xb = x + (size_t)n * 3 * HW;
  float px[3] = { xb[pix], xb[HW + pix], xb[2 * HW + pix] };
  int i0[3]; float fr[3];
#pragma unroll
  for (int c = 0; c < 3; ++c) {
    float xs = fminf(fmaxf(px[c], 0.0f), 1.0f) * (float)(D - 1);
    int i = (int)floorf(xs);
    i = i < 0 ? 0 : (i > D - 2 ? D - 2 : i);
    i0[c] = i; fr[c] = xs - (float)i;
  }
  const float* lut = fused + (size_t)n * 3 * D * D * D;
  const int D3 = D * D * D;
  float o0 = 0.0f, o1 = 0.0f, o2 = 0.0f;
#pragma unroll
  for (int db = 0; db < 2; ++db) {
    const float wb = db ? fr[2] : 1.0f - fr[2];
#pragma unroll
    for (int dg = 0; dg < 2; ++dg) {
      const float wg = dg ? fr[1] : 1.0f - fr[1];
#pragma unroll
      for (int dr = 0; dr < 2; ++dr) {
        const float wr = dr ? fr[0] : 1.0f - fr[0];
        const float wgt = wb * wg * wr;
        const int li = ((i0[2] + db) * D + (i0[1] + dg)) * D + (i0[0] + dr);
        o0 += lut[li] * wgt;
        o1 += lut[D3 + li] * wgt;
        o2 += lut[2 * D3 + li] * wgt;
      }
    }
  }
  const float a = 1.0f / (1.0f + __expf(-logit[n]));
  float* ob = out + (size_t)n * 3 * HW;
  ob[pix]          = fminf(fmaxf((1.0f - a) * px[0] + a * o0, 0.0f), 1.0f);
  ob[HW + pix]     = fminf(fmaxf((1.0f - a) * px[1] + a * o1, 0.0f), 1.0f);
  ob[2 * HW + pix] = fminf(fmaxf((1.0f - a) * px[2] + a * o2, 0.0f), 1.0f);
}

// ---------------------------------------------------------------------------
// host-side orchestration
// ---------------------------------------------------------------------------
extern "C" void kernel_launch(void* const* d_in, const int* in_sizes, int n_in,
                              void* d_out, int out_size, void* d_ws, size_t ws_size,
                              hipStream_t stream) {
  (void)in_sizes; (void)n_in; (void)out_size; (void)ws_size;

  const float* x    = (const float*)d_in[0];
  const float* luts = (const float*)d_in[1];
  const float* cw[8] = { (const float*)d_in[2],  (const float*)d_in[4],
                         (const float*)d_in[6],  (const float*)d_in[8],
                         (const float*)d_in[14], (const float*)d_in[16],
                         (const float*)d_in[18], (const float*)d_in[20] };
  const float* cb[8] = { (const float*)d_in[3],  (const float*)d_in[5],
                         (const float*)d_in[7],  (const float*)d_in[9],
                         (const float*)d_in[15], (const float*)d_in[17],
                         (const float*)d_in[19], (const float*)d_in[21] };
  const float* wn_fw1 = (const float*)d_in[10]; const float* wn_fb1 = (const float*)d_in[11];
  const float* wn_fw2 = (const float*)d_in[12]; const float* wn_fb2 = (const float*)d_in[13];
  const float* bl_fw1 = (const float*)d_in[22]; const float* bl_fb1 = (const float*)d_in[23];
  const float* bl_fw2 = (const float*)d_in[24]; const float* bl_fb2 = (const float*)d_in[25];

  float* out = (float*)d_out;

  const int B = 4, H = 1024, W = 1024, NB = 8, D = 33;
  const int HW = H * W;

  // d_out layout: out(4,3,1024,1024), fused(4,3,33,33,33), tv, mn, w3d(4,8)
  const size_t OUT_FUSED = (size_t)B * 3 * HW;
  const size_t OUT_TV    = OUT_FUSED + (size_t)B * 3 * D * D * D;
  const size_t OUT_MN    = OUT_TV + 1;
  const size_t OUT_W3D   = OUT_MN + 1;

  // per-layer params: {CinReal, Cpad, Cout, Hin, stride}
  const int CinR_[8] = { 3, 32, 32, 64,   3, 16, 16, 32 };
  const int Cpad_[8] = { 4, 32, 32, 64,   4, 16, 16, 32 };
  const int Cout_[8] = { 32, 32, 64, 64, 16, 16, 32, 32 };
  const int Hin_[8]  = { 1024, 1024, 512, 256, 1024, 1024, 512, 256 };
  const int Str_[8]  = { 1, 2, 2, 2,  1, 2, 2, 2 };

  // workspace layout (256-byte aligned)
  char* ws = (char*)d_ws;
  size_t off = 0;
  auto alloc = [&](size_t bytes) -> size_t {
    size_t o = off;
    off = (off + bytes + 255) & ~(size_t)255;
    return o;
  };
  const size_t XH   = alloc((size_t)B * (H + 2) * (W + 2) * 4 * sizeof(f16));   // 33.7 MB
  const size_t ACT0 = alloc((size_t)B * 1026 * 1026 * 32 * sizeof(f16));        // 269.5 MB
  const size_t ACT1 = alloc((size_t)B * 514 * 514 * 32 * sizeof(f16));          // 67.7 MB
  size_t WPK[8];
  int Kpad_[8];
  for (int l = 0; l < 8; ++l) {
    int Ktot = 9 * Cpad_[l];
    Kpad_[l] = ((Ktot + 31) / 32) * 32;
    WPK[l] = alloc((size_t)Cout_[l] * Kpad_[l] * sizeof(f16));
  }
  const size_t FEAT_WN = alloc(B * 64 * sizeof(float));
  const size_t FC1_WN  = alloc(B * 64 * sizeof(float));
  const size_t FEAT_BL = alloc(B * 32 * sizeof(float));
  const size_t FC1_BL  = alloc(B * 32 * sizeof(float));
  const size_t LOGIT   = alloc(B * sizeof(float));

  // 1) x -> NHWC(C=4) f16 with zero halo
  {
    int n = B * (H + 2) * (W + 2);
    pack_x_kernel<<<(n + 255) / 256, 256, 0, stream>>>(x, (f16*)(ws + XH), B, H, W);
  }

  // 2) pack all conv weights
  for (int l = 0; l < 8; ++l) {
    int total = Cout_[l] * Kpad_[l];
    pack_w_kernel<<<(total + 255) / 256, 256, 0, stream>>>(
        cw[l], (f16*)(ws + WPK[l]), Cout_[l], CinR_[l], Cpad_[l], Kpad_[l]);
  }

  // conv launcher helpers
  auto pre_halo = [&](size_t out_off, int l) {
    const int Hout = Hin_[l] / Str_[l];  // strides are 1 or 2 with even dims
    const int n = B * (Hout + 2) * (Hout + 2);
    zero_halo_kernel<<<(n + 255) / 256, 256, 0, stream>>>(
        (f16*)(ws + out_off), B, Hout, Hout, Cout_[l]);
  };
  auto conv_blocks = [&](int l) {
    const int Hout = Hin_[l] / Str_[l];
    const int Mtiles = B * Hout * (Hout >> 4);
    return (Mtiles + 3) / 4;
  };

#define RUN_CONV(l, CI, NT, ST, IN_OFF, OUT_OFF)                               \
  do {                                                                         \
    pre_halo(OUT_OFF, l);                                                      \
    conv3x3_wmma<CI, NT, ST><<<conv_blocks(l), 128, 0, stream>>>(              \
        (const f16*)(ws + (IN_OFF)), (const f16*)(ws + WPK[l]), cb[l],         \
        (f16*)(ws + (OUT_OFF)), B, Hin_[l], Hin_[l]);                          \
  } while (0)

  // 3) weight-net conv stack: XH -> ACT0 -> ACT1 -> ACT0 -> ACT1
  RUN_CONV(0, 4, 2, 1, XH, ACT0);    // (4,1024,1024,32)
  RUN_CONV(1, 32, 2, 2, ACT0, ACT1); // (4,512,512,32)
  RUN_CONV(2, 32, 4, 2, ACT1, ACT0); // (4,256,256,64)
  RUN_CONV(3, 64, 4, 2, ACT0, ACT1); // (4,128,128,64)

  // 4) weight-net head: pool -> fc(relu) -> fc -> d_out w3d
  zero_f32_kernel<<<1, 256, 0, stream>>>((float*)(ws + FEAT_WN), B * 64);
  pool_rows_kernel<<<B * 128, 64, 0, stream>>>((const f16*)(ws + ACT1),
                                               (float*)(ws + FEAT_WN), 128, 128, 64);
  fc_kernel<<<1, 256, 0, stream>>>((const float*)(ws + FEAT_WN), wn_fw1, wn_fb1,
                                   (float*)(ws + FC1_WN), B, 64, 64, 1,
                                   1.0f / (128.0f * 128.0f));
  fc_kernel<<<1, 256, 0, stream>>>((const float*)(ws + FC1_WN), wn_fw2, wn_fb2,
                                   out + OUT_W3D, B, 64, NB, 0, 1.0f);

  // 5) blend-net conv stack
  RUN_CONV(4, 4, 1, 1, XH, ACT0);    // (4,1024,1024,16)
  RUN_CONV(5, 16, 1, 2, ACT0, ACT1); // (4,512,512,16)
  RUN_CONV(6, 16, 2, 2, ACT1, ACT0); // (4,256,256,32)
  RUN_CONV(7, 32, 2, 2, ACT0, ACT1); // (4,128,128,32)

  // 6) blend-net head -> per-image logit
  zero_f32_kernel<<<1, 256, 0, stream>>>((float*)(ws + FEAT_BL), B * 32);
  pool_rows_kernel<<<B * 128, 32, 0, stream>>>((const f16*)(ws + ACT1),
                                               (float*)(ws + FEAT_BL), 128, 128, 32);
  fc_kernel<<<1, 256, 0, stream>>>((const float*)(ws + FEAT_BL), bl_fw1, bl_fb1,
                                   (float*)(ws + FC1_BL), B, 32, 32, 1,
                                   1.0f / (128.0f * 128.0f));
  fc_kernel<<<1, 256, 0, stream>>>((const float*)(ws + FC1_BL), bl_fw2, bl_fb2,
                                   (float*)(ws + LOGIT), B, 32, 1, 0, 1.0f);

  // 7) fused LUTs (into d_out, read back by later kernels)
  {
    int total = B * 3 * D * D * D;
    fuse_luts_kernel<<<(total + 255) / 256, 256, 0, stream>>>(
        out + OUT_W3D, luts, out + OUT_FUSED, B, NB, D);
  }

  // 8) TV + monotonicity regularizers
  zero_f32_kernel<<<1, 32, 0, stream>>>(out + OUT_TV, 2);
  tv_mn_kernel<<<256, 256, 0, stream>>>(out + OUT_FUSED, out + OUT_TV,
                                        out + OUT_MN, B, D);

  // 9) trilinear LUT apply + blend -> final image
  {
    int total = B * HW;
    trilerp_blend_kernel<<<(total + 255) / 256, 256, 0, stream>>>(
        x, out + OUT_FUSED, (const float*)(ws + LOGIT), out, B, HW, D);
  }
#undef RUN_CONV
}
